// CausalSelfAttention_54185307406548
// MI455X (gfx1250) — compile-verified, matrix operand order checked by reading, not run on hardware
//
#include <hip/hip_runtime.h>
#include <hip/hip_bf16.h>
#include <math.h>

typedef __attribute__((ext_vector_type(16))) _Float16 v16h;
typedef __attribute__((ext_vector_type(8)))  _Float16 v8h;
typedef __attribute__((ext_vector_type(4)))  _Float16 v4h;
typedef __attribute__((ext_vector_type(8)))  float    v8f;

// ---------------------------------------------------------------------------
// gfx1250 async global->LDS copy (16B per lane), tracked by ASYNCcnt.
// ---------------------------------------------------------------------------
__device__ __forceinline__ void async_copy_b128(unsigned lds_off, const _Float16* g) {
  asm volatile("global_load_async_to_lds_b128 %0, %1, off"
               :: "v"(lds_off), "v"((unsigned long long)(uintptr_t)g)
               : "memory");
}
__device__ __forceinline__ void wait_async0() {
  asm volatile("s_wait_asynccnt 0x0" ::: "memory");
}
__device__ __forceinline__ void wait_ds0() {   // wave-local LDS store->load order
  asm volatile("s_wait_dscnt 0x0" ::: "memory");
}
__device__ __forceinline__ unsigned lds_off_of(const void* p) {
  return (unsigned)(uintptr_t)p;   // flat LDS ptr low 32 bits == LDS offset
}

// ---------------------------------------------------------------------------
// WMMA fragment loader (wave32). K-contiguous tiles, A- and B-operand layouts
// coincide: lane L -> row/col = L&15, half = L>>4;
// elements [0..7] = K = half*8+0..7 ; [8..15] = K = 16+half*8+0..7.
// ---------------------------------------------------------------------------
__device__ __forceinline__ v16h load_frag16(const _Float16* base, int stride_h) {
  const int lane = threadIdx.x & 31;
  const int r    = lane & 15;
  const int hf8  = (lane >> 4) << 3;
  const _Float16* p = base + r * stride_h + hf8;
  v8h lo = *(const v8h*)(p);
  v8h hi = *(const v8h*)(p + 16);
  return __builtin_shufflevector(lo, hi, 0,1,2,3,4,5,6,7,8,9,10,11,12,13,14,15);
}

__device__ __forceinline__ v8f wmma_f16(v16h a, v16h b, v8f c) {
  return __builtin_amdgcn_wmma_f32_16x16x32_f16(false, a, false, b, (short)0, c,
                                                false, false);
}

// ---------------------------------------------------------------------------
// f32 -> f16 conversion (one-time)
// ---------------------------------------------------------------------------
__global__ __launch_bounds__(256)
void cvt_f32_to_f16_kernel(const float* __restrict__ in, _Float16* __restrict__ out,
                           int n4) {
  const int i = blockIdx.x * 256 + threadIdx.x;
  if (i < n4) {
    const float4 v = ((const float4*)in)[i];
    v4h h;
    h[0] = (_Float16)v.x; h[1] = (_Float16)v.y;
    h[2] = (_Float16)v.z; h[3] = (_Float16)v.w;
    ((v4h*)out)[i] = h;
  }
}

// ---------------------------------------------------------------------------
// GEMM: Y[M,N] = A[M,K] @ W[N,K]^T + bias[N], f16 in, f16/f32 out.
// 128x128x64 block tile, 8 waves, double-buffered async LDS staging.
// ---------------------------------------------------------------------------
#define GB_BM 128
#define GB_BN 128
#define GB_BK 64
#define GB_STR (GB_BK + 8)   // 72 halfs = 144B rows, 16B aligned

template <bool OUT_F16>
__global__ __launch_bounds__(256)
void gemm_bias_wmma_kernel(const _Float16* __restrict__ A,
                           const _Float16* __restrict__ W,
                           const float* __restrict__ bias,
                           void* __restrict__ Yv,
                           int M, int N, int K) {
  __shared__ _Float16 sA[2][GB_BM][GB_STR];
  __shared__ _Float16 sB[2][GB_BN][GB_STR];

  const int tid  = threadIdx.x;
  const int lane = tid & 31;
  const int wave = tid >> 5;
  const int wm   = (wave & 3) * 32;
  const int wn   = (wave >> 2) * 64;
  const int bm   = blockIdx.y * GB_BM;
  const int bn   = blockIdx.x * GB_BN;

  int crow[4], ccol[4];
#pragma unroll
  for (int i = 0; i < 4; ++i) {
    const int chunk = tid + i * 256;
    crow[i] = chunk >> 3;
    ccol[i] = (chunk & 7) * 8;
  }

  v8f acc[2][4] = {};
  const int nK = K / GB_BK;

  // Prologue: stage tile 0 into buffer 0.
#pragma unroll
  for (int i = 0; i < 4; ++i) {
    async_copy_b128(lds_off_of(&sA[0][crow[i]][ccol[i]]),
                    A + (size_t)(bm + crow[i]) * K + ccol[i]);
    async_copy_b128(lds_off_of(&sB[0][crow[i]][ccol[i]]),
                    W + (size_t)(bn + crow[i]) * K + ccol[i]);
  }
  wait_async0();
  __syncthreads();

  for (int kt = 0; kt < nK; ++kt) {
    const int cur = kt & 1;
    const int nxt = cur ^ 1;
    if (kt + 1 < nK) {          // async copy of next tile overlaps WMMA below
      const int k0n = (kt + 1) * GB_BK;
#pragma unroll
      for (int i = 0; i < 4; ++i) {
        async_copy_b128(lds_off_of(&sA[nxt][crow[i]][ccol[i]]),
                        A + (size_t)(bm + crow[i]) * K + k0n + ccol[i]);
        async_copy_b128(lds_off_of(&sB[nxt][crow[i]][ccol[i]]),
                        W + (size_t)(bn + crow[i]) * K + k0n + ccol[i]);
      }
    }
#pragma unroll
    for (int kk = 0; kk < 2; ++kk) {
      v16h af0 = load_frag16(&sA[cur][wm][kk * 32],      GB_STR);
      v16h af1 = load_frag16(&sA[cur][wm + 16][kk * 32], GB_STR);
#pragma unroll
      for (int nt = 0; nt < 4; ++nt) {
        v16h bf = load_frag16(&sB[cur][wn + nt * 16][kk * 32], GB_STR);
        acc[0][nt] = wmma_f16(af0, bf, acc[0][nt]);
        acc[1][nt] = wmma_f16(af1, bf, acc[1][nt]);
      }
    }
    wait_async0();
    __syncthreads();
  }

  const int r16 = lane & 15;
  const int hf  = lane >> 4;
#pragma unroll
  for (int nt = 0; nt < 4; ++nt) {
    const int gn = bn + wn + nt * 16 + r16;
    const float bv = bias[gn];
#pragma unroll
    for (int mt = 0; mt < 2; ++mt) {
#pragma unroll
      for (int r = 0; r < 8; ++r) {
        const int gm = bm + wm + mt * 16 + hf * 8 + r;
        const float val = acc[mt][nt][r] + bv;
        if (OUT_F16)
          ((_Float16*)Yv)[(size_t)gm * N + gn] = (_Float16)val;
        else
          ((float*)Yv)[(size_t)gm * N + gn] = val;
      }
    }
  }
}

// ---------------------------------------------------------------------------
// Flash attention (causal), f16 in/out. Register-resident online softmax:
// row statistics live in the WMMA C-layout element slots (row = half*8+r),
// cross-lane reductions via shfl_xor over the 16-lane half-groups.
// Block = (b, h, 128 query rows); 8 waves, 16 rows each; 64-key blocks.
// ---------------------------------------------------------------------------
#define AT_T  2048
#define AT_C  1024
#define AT_C3 3072
#define AT_H  16
#define AT_D  64
#define AT_QSTR (AT_D + 8)   // 72
#define AT_PSTR (64 + 8)     // 72

__global__ __launch_bounds__(256)
void flash_attn_wmma_kernel(const _Float16* __restrict__ qkv,
                            _Float16* __restrict__ y) {
  __shared__ _Float16 sQ[128][AT_QSTR];
  __shared__ _Float16 sK[64][AT_QSTR];
  __shared__ _Float16 sVt[AT_D][64 + 8];     // transposed: sVt[d][key]
  __shared__ _Float16 sP[8][16][AT_PSTR];    // per-wave probabilities (A-layout)

  const int tid  = threadIdx.x;
  const int lane = tid & 31;
  const int wave = tid >> 5;
  const int r16  = lane & 15;
  const int hf   = lane >> 4;

  const int blk   = blockIdx.x;
  const int qb    = blk & 15;
  const int h     = (blk >> 4) & 15;
  const int b     = blk >> 8;
  const int qbase = qb * 128;
  const _Float16* base = qkv + (size_t)b * AT_T * AT_C3;

  // Stage Q async: 1024 chunks of 16B, 4/thread.
#pragma unroll
  for (int i = 0; i < 4; ++i) {
    const int chunk = tid + i * 256;
    const int row = chunk >> 3;
    const int c8  = (chunk & 7) * 8;
    async_copy_b128(lds_off_of(&sQ[row][c8]),
                    base + (size_t)(qbase + row) * AT_C3 + h * AT_D + c8);
  }
  wait_async0();
  __syncthreads();

  const int m0 = wave * 16;
  const v16h qf0 = load_frag16(&sQ[m0][0],  AT_QSTR);
  const v16h qf1 = load_frag16(&sQ[m0][32], AT_QSTR);

  v8f o[4] = {};
  v8f m_i, l_i;
#pragma unroll
  for (int r = 0; r < 8; ++r) { m_i[r] = -__builtin_inff(); l_i[r] = 0.f; }
  const float scale = 0.125f;     // 1/sqrt(64)
  const float L2E   = 1.4426950408889634f;
  const int   qtop  = qbase + m0 + 15;   // this wave's last (largest) query row

  const int nkb = (qbase + 128) / 64;
  for (int kb = 0; kb < nkb; ++kb) {
    const int ks = kb * 64;
    // K tile: async (512 chunks, 2/thread). V tile: regs -> transposed LDS.
#pragma unroll
    for (int i = 0; i < 2; ++i) {
      const int chunk = tid + i * 256;
      const int row = chunk >> 3;
      const int c8  = (chunk & 7) * 8;
      async_copy_b128(lds_off_of(&sK[row][c8]),
                      base + (size_t)(ks + row) * AT_C3 + AT_C + h * AT_D + c8);
    }
#pragma unroll
    for (int i = 0; i < 2; ++i) {
      const int chunk = tid + i * 256;
      const int row = chunk >> 3;
      const int c8  = (chunk & 7) * 8;
      const v8h vv = *(const v8h*)(base + (size_t)(ks + row) * AT_C3 + 2 * AT_C + h * AT_D + c8);
#pragma unroll
      for (int j = 0; j < 8; ++j) sVt[c8 + j][row] = vv[j];
    }
    wait_async0();
    __syncthreads();

    // Waves whose rows are entirely above this key block skip (wave-uniform).
    if (ks <= qtop) {
      // S = scale * Q @ K^T  (16x64), causal mask in registers.
      v8f s[4];
      const bool diag = (ks + 63 > qbase + m0);   // block touches the diagonal
#pragma unroll
      for (int nt = 0; nt < 4; ++nt) {
        v16h kf0 = load_frag16(&sK[nt * 16][0],  AT_QSTR);
        v16h kf1 = load_frag16(&sK[nt * 16][32], AT_QSTR);
        v8f t = {};
        t = wmma_f16(qf0, kf0, t);
        t = wmma_f16(qf1, kf1, t);
        if (diag) {
          const int kcol = ks + nt * 16 + r16;
#pragma unroll
          for (int r = 0; r < 8; ++r) {
            const int qrow = qbase + m0 + hf * 8 + r;
            s[nt][r] = (kcol <= qrow) ? t[r] * scale : -__builtin_inff();
          }
        } else {
#pragma unroll
          for (int r = 0; r < 8; ++r) s[nt][r] = t[r] * scale;
        }
      }

      // Row max across the 4 tiles, then across the 16-lane half-group.
      v8f smax;
#pragma unroll
      for (int r = 0; r < 8; ++r)
        smax[r] = fmaxf(fmaxf(s[0][r], s[1][r]), fmaxf(s[2][r], s[3][r]));
#pragma unroll
      for (int off = 1; off < 16; off <<= 1)
#pragma unroll
        for (int r = 0; r < 8; ++r)
          smax[r] = fmaxf(smax[r], __shfl_xor(smax[r], off));

      v8f newm, alpha, lsum;
#pragma unroll
      for (int r = 0; r < 8; ++r) {
        newm[r]  = fmaxf(m_i[r], smax[r]);
        alpha[r] = exp2f((m_i[r] - newm[r]) * L2E);
        lsum[r]  = 0.f;
      }
      // P = exp(S - m): store f16 into sP (this is the C->A layout transpose).
#pragma unroll
      for (int nt = 0; nt < 4; ++nt)
#pragma unroll
        for (int r = 0; r < 8; ++r) {
          const float p = exp2f((s[nt][r] - newm[r]) * L2E);
          sP[wave][hf * 8 + r][nt * 16 + r16] = (_Float16)p;
          lsum[r] += p;
        }
#pragma unroll
      for (int off = 1; off < 16; off <<= 1)
#pragma unroll
        for (int r = 0; r < 8; ++r) lsum[r] += __shfl_xor(lsum[r], off);
#pragma unroll
      for (int r = 0; r < 8; ++r) {
        l_i[r] = alpha[r] * l_i[r] + lsum[r];
        m_i[r] = newm[r];
      }
      // Rescale running O, then O += P @ V (sP is wave-local; LDS in-order).
#pragma unroll
      for (int nt = 0; nt < 4; ++nt)
#pragma unroll
        for (int r = 0; r < 8; ++r) o[nt][r] *= alpha[r];

      wait_ds0();
      v16h pf0 = load_frag16(&sP[wave][0][0],  AT_PSTR);
      v16h pf1 = load_frag16(&sP[wave][0][32], AT_PSTR);
#pragma unroll
      for (int nt = 0; nt < 4; ++nt) {
        v16h vf0 = load_frag16(&sVt[nt * 16][0],  AT_PSTR);  // keys 0..31
        v16h vf1 = load_frag16(&sVt[nt * 16][32], AT_PSTR);  // keys 32..63
        o[nt] = wmma_f16(pf0, vf0, o[nt]);
        o[nt] = wmma_f16(pf1, vf1, o[nt]);
      }
    }
    __syncthreads();   // before next block overwrites sK/sVt
  }

  // Normalize (l lives in the matching element slots) and write y (f16).
  _Float16* yb = y + (size_t)b * AT_T * AT_C;
#pragma unroll
  for (int nt = 0; nt < 4; ++nt) {
#pragma unroll
    for (int r = 0; r < 8; ++r) {
      const int row = m0 + hf * 8 + r;
      yb[(size_t)(qbase + row) * AT_C + h * AT_D + nt * 16 + r16] =
          (_Float16)(o[nt][r] / l_i[r]);
    }
  }
}

// ---------------------------------------------------------------------------
extern "C" void kernel_launch(void* const* d_in, const int* in_sizes, int n_in,
                              void* d_out, int out_size, void* d_ws, size_t ws_size,
                              hipStream_t stream) {
  const float* x      = (const float*)d_in[0];
  const float* W_attn = (const float*)d_in[1];
  const float* b_attn = (const float*)d_in[2];
  const float* W_proj = (const float*)d_in[3];
  const float* b_proj = (const float*)d_in[4];
  float* out = (float*)d_out;

  const int B = 4, T = 2048, C = 1024;
  const int M = B * T;                         // 8192

  // Workspace (all f16): xh | Wah | Wph | qkvh | yh
  _Float16* xh   = (_Float16*)d_ws;            // M*C
  _Float16* Wah  = xh   + (size_t)M * C;       // 3C*C
  _Float16* Wph  = Wah  + (size_t)3 * C * C;   // C*C
  _Float16* qkvh = Wph  + (size_t)C * C;       // M*3C
  _Float16* yh   = qkvh + (size_t)M * 3 * C;   // M*C

  dim3 blk(256);

  // 0) one-time f32 -> f16 conversions
  {
    int n4 = (M * C) / 4;
    cvt_f32_to_f16_kernel<<<dim3((n4 + 255) / 256), blk, 0, stream>>>(x, xh, n4);
    n4 = (3 * C * C) / 4;
    cvt_f32_to_f16_kernel<<<dim3((n4 + 255) / 256), blk, 0, stream>>>(W_attn, Wah, n4);
    n4 = (C * C) / 4;
    cvt_f32_to_f16_kernel<<<dim3((n4 + 255) / 256), blk, 0, stream>>>(W_proj, Wph, n4);
  }

  // 1) qkv = x @ W_attn^T + b_attn     (8192 x 3072 x 1024, f16 out)
  gemm_bias_wmma_kernel<true><<<dim3(3 * C / GB_BN, M / GB_BM), blk, 0, stream>>>(
      xh, Wah, b_attn, qkvh, M, 3 * C, C);

  // 2) flash attention -> yh           (B*H*(T/128) = 1024 blocks)
  flash_attn_wmma_kernel<<<dim3(B * AT_H * (T / 128)), blk, 0, stream>>>(qkvh, yh);

  // 3) out = yh @ W_proj^T + b_proj    (8192 x 1024 x 1024, f32 out)
  gemm_bias_wmma_kernel<false><<<dim3(C / GB_BN, M / GB_BM), blk, 0, stream>>>(
      yh, Wph, b_proj, out, M, C, C);
}
